// NTXentLoss_37984690766093
// MI455X (gfx1250) — matrix operand order repaired and checked by reading, not compiled, not run on hardware
//
#include <hip/hip_runtime.h>
#include <hip/hip_bf16.h>

typedef __attribute__((ext_vector_type(16))) _Float16 v16h;
typedef __attribute__((ext_vector_type(8)))  float    v8f;

#define D            256
#define NROWS        8192
#define HALF_N       4096
#define ROWS_PER_WG  32
#define WAVES_PER_WG 8
#define NTILES       (NROWS / 16)   // 512 column tiles

union V16H { v16h v; uint4 q[2]; };
union H8   { uint4 q; _Float16 h[8]; };

// ---------------------------------------------------------------------------
// Phase 1: row-normalize concat(z1,z2) in fp32, emit fp16 zn into workspace.
// One block per row, 256 threads = 256 elements.
// ---------------------------------------------------------------------------
__global__ void __launch_bounds__(256)
ntxent_normalize(const float* __restrict__ z1, const float* __restrict__ z2,
                 _Float16* __restrict__ zn) {
  const int row = blockIdx.x;
  const int t   = threadIdx.x;
  const float* src = (row < HALF_N) ? (z1 + (size_t)row * D)
                                    : (z2 + (size_t)(row - HALF_N) * D);
  float x = src[t];
  __shared__ float sd[256];
  sd[t] = x * x;
  __syncthreads();
  for (int off = 128; off > 0; off >>= 1) {
    if (t < off) sd[t] += sd[t + off];
    __syncthreads();
  }
  float nrm = fmaxf(sqrtf(sd[0]), 1e-8f);
  zn[(size_t)row * D + t] = (_Float16)(x / nrm);
}

// ---------------------------------------------------------------------------
// Phase 2: fused  sim = (zn zn^T)*2  +  per-row sum(exp)  over ALL columns
// (diagonal included — corrected in the tail), then cross-wave combine and
// per-row  log(S - exp(self)) - pos.
// Block = 32 rows (2 WMMA M-tiles), 8 waves split 512 column tiles.
// Hot loop is a single basic block: 16 b128 loads + 16 WMMAs + 16 exp-adds.
// ---------------------------------------------------------------------------
__global__ void __launch_bounds__(256)
ntxent_simloss(const _Float16* __restrict__ zn, float* __restrict__ partial) {
  const int lane = threadIdx.x & 31;
  const int wave = threadIdx.x >> 5;
  const int hi   = lane >> 4;     // half-wave select
  const int lo   = lane & 15;
  const int rowBase0 = blockIdx.x * ROWS_PER_WG;
  const int rowBase1 = rowBase0 + 16;

  // A fragments: ISA 16-bit A 16x32 layout.
  // lane L (L<16): row M=L, K = {k0..k0+7, k0+16..k0+23}
  // lane L+16:     row M=L, K = {k0+8..k0+15, k0+24..k0+31}
  v16h a0[8], a1[8];
  {
    const _Float16* ap0 = zn + (size_t)(rowBase0 + lo) * D + hi * 8;
    const _Float16* ap1 = zn + (size_t)(rowBase1 + lo) * D + hi * 8;
#pragma unroll
    for (int kc = 0; kc < 8; ++kc) {
      V16H t0, t1;
      t0.q[0] = *(const uint4*)(ap0 + kc * 32);
      t0.q[1] = *(const uint4*)(ap0 + kc * 32 + 16);
      t1.q[0] = *(const uint4*)(ap1 + kc * 32);
      t1.q[1] = *(const uint4*)(ap1 + kc * 32 + 16);
      a0[kc] = t0.v;
      a1[kc] = t1.v;
    }
  }

  float s[16];                       // running sum(exp) per owned row (this wave's tiles)
#pragma unroll
  for (int r = 0; r < 16; ++r) s[r] = 0.0f;

  for (int j = wave; j < NTILES; j += WAVES_PER_WG) {
    const int col = j * 16 + lo;
    // B fragment: lane L holds column N=L&15, K = {k0+(L>>4)*8..+7, k0+16+(L>>4)*8..+7} (matches A)
    const _Float16* bp = zn + (size_t)col * D + hi * 8;
    v8f c0 = {};
    v8f c1 = {};
#pragma unroll
    for (int kc = 0; kc < 8; ++kc) {
      V16H b;
      b.q[0] = *(const uint4*)(bp + kc * 32);
      b.q[1] = *(const uint4*)(bp + kc * 32 + 16);
      c0 = __builtin_amdgcn_wmma_f32_16x16x32_f16(false, a0[kc], false, b.v,
                                                  (short)0, c0, false, false);
      c1 = __builtin_amdgcn_wmma_f32_16x16x32_f16(false, a1[kc], false, b.v,
                                                  (short)0, c1, false, false);
    }
    // Lean epilogue (no masks, no branches): diagonal handled in the tail.
    // C/D layout — VGPR r, lanes<16: M=r, lanes>=16: M=r+8, N=lane&15
#pragma unroll
    for (int r = 0; r < 8; ++r) {
      s[r]     += __expf(2.0f * c0[r]);
      s[8 + r] += __expf(2.0f * c1[r]);
    }
  }

  // Per-row totals within the wave: a C row lives in 16 lanes of one half-wave.
#pragma unroll
  for (int m = 1; m <= 8; m <<= 1) {
#pragma unroll
    for (int r = 0; r < 16; ++r) s[r] += __shfl_xor(s[r], m, 16);
  }

  // Stage per-wave per-row partials: sm[wave][rowLocal], rowLocal 0..31.
  __shared__ float sm[WAVES_PER_WG][ROWS_PER_WG + 1];
  if (lo == 0) {
#pragma unroll
    for (int r = 0; r < 8; ++r) {
      sm[wave][hi * 8 + r]      = s[r];       // tile0 rows
      sm[wave][16 + hi * 8 + r] = s[8 + r];   // tile1 rows
    }
  }
  __syncthreads();

  // Tail: 32 threads, one per owned row. Deterministic cross-wave sum,
  // then exact-ish corrections recomputed as fp32 dots over f16 data.
  __shared__ float red[ROWS_PER_WG];
  if (threadIdx.x < ROWS_PER_WG) {
    const int t   = threadIdx.x;
    const int row = rowBase0 + t;              // rowLocal t -> global row
    float S = 0.0f;
#pragma unroll
    for (int w = 0; w < WAVES_PER_WG; ++w) S += sm[w][t];

    const _Float16* rp = zn + (size_t)row * D;
    const _Float16* mp = zn + (size_t)(row ^ HALF_N) * D;
    float selfdot = 0.0f, posdot = 0.0f;
    for (int kc = 0; kc < D / 8; ++kc) {
      H8 a, b;
      a.q = *(const uint4*)(rp + kc * 8);
      b.q = *(const uint4*)(mp + kc * 8);
#pragma unroll
      for (int e = 0; e < 8; ++e) {
        float av = (float)a.h[e];
        float bv = (float)b.h[e];
        selfdot += av * av;
        posdot  += av * bv;
      }
    }
    // Remove diagonal term from the denominator, subtract positive logit.
    red[t] = __logf(S - __expf(2.0f * selfdot)) - 2.0f * posdot;
  }
  __syncthreads();
  if (threadIdx.x == 0) {
    float acc = 0.0f;
    for (int i = 0; i < ROWS_PER_WG; ++i) acc += red[i];
    partial[blockIdx.x] = acc;
  }
}

// ---------------------------------------------------------------------------
// Phase 3: deterministic final reduction of per-block partials -> mean loss.
// ---------------------------------------------------------------------------
__global__ void __launch_bounds__(256)
ntxent_finalize(const float* __restrict__ partial, float* __restrict__ out, int n) {
  __shared__ float sd[256];
  const int t = threadIdx.x;
  sd[t] = (t < n) ? partial[t] : 0.0f;
  __syncthreads();
  for (int off = 128; off > 0; off >>= 1) {
    if (t < off) sd[t] += sd[t + off];
    __syncthreads();
  }
  if (t == 0) out[0] = sd[0] * (1.0f / NROWS);
}

extern "C" void kernel_launch(void* const* d_in, const int* in_sizes, int n_in,
                              void* d_out, int out_size, void* d_ws, size_t ws_size,
                              hipStream_t stream) {
  const float* z1 = (const float*)d_in[0];
  const float* z2 = (const float*)d_in[1];
  _Float16* zn      = (_Float16*)d_ws;
  float*    partial = (float*)((char*)d_ws + (size_t)NROWS * D * sizeof(_Float16));
  float*    out     = (float*)d_out;

  ntxent_normalize<<<NROWS, 256, 0, stream>>>(z1, z2, zn);
  const int nblk = NROWS / ROWS_PER_WG;  // 256
  ntxent_simloss<<<nblk, 256, 0, stream>>>(zn, partial);
  ntxent_finalize<<<1, 256, 0, stream>>>(partial, out, nblk);
}
